// TemporalFlowCell_87497073754706
// MI455X (gfx1250) — compile-verified, matrix operand order checked by reading, not run on hardware
//
#include <hip/hip_runtime.h>
#include <hip/hip_bf16.h>
#include <stddef.h>

// ---------------------------------------------------------------------------
// TemporalFlowCell on MI455X (gfx1250, wave32).
//   B=8, S=4096, D=1024, K=128.
//   GEMMs: v_wmma_f32_16x16x32_bf16, barrier-free / LDS-free. Each wave owns a
//   16-row M-subtile; A fragments are built in registers with native packed
//   f32->bf16 converts; B comes pre-packed in the exact B-operand lane layout
//   (two global_load_b128 per fragment). Loads are batched ahead of the WMMA
//   chain so B-fetch latency hides under matrix math.
//   The serial recurrence runs in fp32 with R cached in registers.
// ---------------------------------------------------------------------------

typedef __attribute__((ext_vector_type(16))) __bf16 v16bf;
typedef __attribute__((ext_vector_type(8)))  __bf16 v8bf;
typedef __attribute__((ext_vector_type(8)))  float  v8f;
typedef __attribute__((ext_vector_type(8)))  float  v8ff;

union ABu {
    v16bf bf;     // 16 bf16 = 32 bytes = 8 VGPRs (WMMA A/B operand)
    v8bf  h[2];
    uint4 u[2];
};

__device__ __forceinline__ unsigned short f2bf(float f) {
    // round-to-nearest-even f32 -> bf16 (bit trick; used only in pack kernel)
    unsigned int u = __float_as_uint(f);
    unsigned int r = u + 0x7FFFu + ((u >> 16) & 1u);
    return (unsigned short)(r >> 16);
}

// ---------------------------------------------------------------------------
// Pack a weight matrix W (N rows x rowstride cols, fp32, row-major) into the
// bf16 B-operand layout for v_wmma_f32_16x16x32_bf16:
//   Bp[((kc*N + n)*2 + half)*16 + e] = bf16( W[n][kc*32 + klocal(e,half)] )
//   klocal: half0 -> {0..7,16..23}, half1 -> {8..15,24..31}
// ---------------------------------------------------------------------------
__global__ void __launch_bounds__(256) pack_b_kernel(
    const float* __restrict__ W, unsigned short* __restrict__ out,
    int N, int rowstride)
{
    int t = blockIdx.x * 256 + threadIdx.x;
    int e    = t & 15;
    int half = (t >> 4) & 1;
    int g    = t >> 5;
    int n    = g % N;
    int kc   = g / N;
    int klocal = (e < 8 ? e : e + 8) + half * 8;
    out[t] = f2bf(W[(size_t)n * rowstride + kc * 32 + klocal]);
}

// ---------------------------------------------------------------------------
// GEMM1: beta[m,k] = sum_d x[m,d] * W_in[k,d]    (M=32768, Kdim=1024, N=128)
// Block = 128 rows (8 waves x 16-row subtiles), full N=128 per wave.
// Per k-step: A f32 loads + 16 B loads issued up front, converts overlap the
// loads, then 8 WMMAs with incremental load waits. No LDS, no barriers.
// ---------------------------------------------------------------------------
__global__ void __launch_bounds__(256) gemm_beta_kernel(
    const float* __restrict__ x, const uint4* __restrict__ Bp,
    float* __restrict__ beta)
{
    const int tid  = threadIdx.x;
    const int wave = tid >> 5;
    const int lane = tid & 31;
    const int half = lane >> 4;
    const int row  = lane & 15;
    const int m0   = blockIdx.x * 128 + wave * 16;     // wave's 16 rows

    const float* xrow = x + (size_t)(m0 + row) * 1024; // this lane's row

    v8f acc[8];
    #pragma unroll
    for (int nt = 0; nt < 8; ++nt)
        acc[nt] = (v8f){0.f,0.f,0.f,0.f,0.f,0.f,0.f,0.f};

    for (int kc = 0; kc < 32; ++kc) {
        // issue A loads first (results needed earliest)
        const v8ff* xw = reinterpret_cast<const v8ff*>(xrow + kc * 32);
        v8ff f0 = xw[half];          // k-local half*8 .. +7
        v8ff f1 = xw[2 + half];      // k-local 16+half*8 .. +7

        // issue all B loads (one clause of 16 global_load_b128)
        ABu b[8];
        #pragma unroll
        for (int nt = 0; nt < 8; ++nt) {
            int ncol = nt * 16 + row;
            const uint4* bp = Bp + ((size_t)((kc * 128 + ncol) * 2 + half)) * 2;
            b[nt].u[0] = bp[0];
            b[nt].u[1] = bp[1];
        }

        // convert A while B loads are in flight (v_cvt_pk_bf16_f32)
        ABu a;
        a.h[0] = __builtin_convertvector(f0, v8bf);
        a.h[1] = __builtin_convertvector(f1, v8bf);

        #pragma unroll
        for (int nt = 0; nt < 8; ++nt)
            acc[nt] = __builtin_amdgcn_wmma_f32_16x16x32_bf16(
                          false, a.bf, false, b[nt].bf, (short)0, acc[nt],
                          false, false);
    }
    #pragma unroll
    for (int nt = 0; nt < 8; ++nt) {
        int ncol = nt * 16 + row;
        #pragma unroll
        for (int r = 0; r < 8; ++r) {
            int gr = m0 + r + half * 8;      // C layout: lanes>=16 hold M+8
            beta[(size_t)gr * 128 + ncol] = acc[nt][r];
        }
    }
}

// ---------------------------------------------------------------------------
// Recurrent scan (fp32). One block per batch, 512 threads:
//   k = tid&127 owns state column k; jg = tid>>7 owns 32 rows of R (in regs).
// ---------------------------------------------------------------------------
__global__ void __launch_bounds__(512) scan_kernel(
    const float* __restrict__ beta, const float* __restrict__ alpha,
    const float* __restrict__ omega, const float* __restrict__ R,
    float* __restrict__ srseq, float* __restrict__ srf, float* __restrict__ sif)
{
    __shared__ float2 nrni[128];
    __shared__ float2 part[4][128];
    const int tid = threadIdx.x;
    const int k2  = tid & 127;
    const int jg  = tid >> 7;
    const int b   = blockIdx.x;

    float r[32];
    #pragma unroll
    for (int jj = 0; jj < 32; ++jj)
        r[jj] = R[(size_t)(jg * 32 + jj) * 128 + k2];

    float mag = 0.f, cw = 0.f, sw = 0.f, sr = 0.f, si = 0.f;
    if (tid < 128) {
        mag = 1.0f / (1.0f + expf(-alpha[k2]));
        float w = omega[k2] * 1.0f;          // DT = 1.0
        cw = cosf(w);
        sw = sinf(w);
    }

    const float* bb = beta  + (size_t)b * 4096 * 128;
    float*       so = srseq + (size_t)b * 4096 * 128;

    for (int t = 0; t < 4096; ++t) {
        if (tid < 128) {
            float bt = bb[(size_t)t * 128 + k2];
            if (t + 1 < 4096)
                __builtin_prefetch(bb + (size_t)(t + 1) * 128 + k2, 0, 0);
            float nr = fmaf(mag, fmaf(sr, cw, -(si * sw)), bt);
            float ni = mag * fmaf(sr, sw, si * cw);
            nrni[k2] = make_float2(nr, ni);
        }
        __syncthreads();
        float psr = 0.f, psi = 0.f;
        #pragma unroll
        for (int jj = 0; jj < 32; ++jj) {
            float2 v = nrni[jg * 32 + jj];   // ds_load_b64 broadcast
            psr = fmaf(v.x, r[jj], psr);
            psi = fmaf(v.y, r[jj], psi);
        }
        part[jg][k2] = make_float2(psr, psi);
        __syncthreads();
        if (tid < 128) {
            float2 p0 = part[0][k2], p1 = part[1][k2];
            float2 p2 = part[2][k2], p3 = part[3][k2];
            sr = (p0.x + p1.x) + (p2.x + p3.x);
            si = (p0.y + p1.y) + (p2.y + p3.y);
            so[(size_t)t * 128 + k2] = sr;
        }
    }
    if (tid < 128) {                         // tuple tail: sr_f, si_f (B,K)
        srf[b * 128 + k2] = sr;
        sif[b * 128 + k2] = si;
    }
}

// ---------------------------------------------------------------------------
// GEMM2: y[m,d] = sum_k srseq[m,k] * W_out[d,k] + b_out[d]
//   (M=32768, Kdim=128, N=1024). Block = 128 rows (8 waves x 16-row subtiles);
// each wave pre-converts its 16x128 A panel into 4 register fragments, then
// sweeps the 64 N-tiles two at a time (16 B loads in flight per step).
// No LDS, no barriers.
// ---------------------------------------------------------------------------
__global__ void __launch_bounds__(256) gemm_out_kernel(
    const float* __restrict__ srseq, const uint4* __restrict__ Bp,
    const float* __restrict__ b_out, float* __restrict__ y)
{
    const int tid  = threadIdx.x;
    const int wave = tid >> 5;
    const int lane = tid & 31;
    const int half = lane >> 4;
    const int row  = lane & 15;
    const int m0   = blockIdx.x * 128 + wave * 16;

    const float* sp = srseq + (size_t)(m0 + row) * 128;  // lane's A row (K=128)

    ABu a[4];
    #pragma unroll
    for (int kc = 0; kc < 4; ++kc) {
        const v8ff* xw = reinterpret_cast<const v8ff*>(sp + kc * 32);
        v8ff f0 = xw[half];
        v8ff f1 = xw[2 + half];
        a[kc].h[0] = __builtin_convertvector(f0, v8bf);
        a[kc].h[1] = __builtin_convertvector(f1, v8bf);
    }

    for (int nt = 0; nt < 64; nt += 2) {
        int ncol0 = nt * 16 + row;
        int ncol1 = ncol0 + 16;

        // issue both tiles' B loads up front (16 global_load_b128)
        ABu b0[4], b1[4];
        #pragma unroll
        for (int kc = 0; kc < 4; ++kc) {
            const uint4* bp0 = Bp + ((size_t)((kc * 1024 + ncol0) * 2 + half)) * 2;
            const uint4* bp1 = Bp + ((size_t)((kc * 1024 + ncol1) * 2 + half)) * 2;
            b0[kc].u[0] = bp0[0];
            b0[kc].u[1] = bp0[1];
            b1[kc].u[0] = bp1[0];
            b1[kc].u[1] = bp1[1];
        }
        float bias0 = b_out[ncol0];
        float bias1 = b_out[ncol1];

        v8f acc0 = (v8f){0.f,0.f,0.f,0.f,0.f,0.f,0.f,0.f};
        v8f acc1 = (v8f){0.f,0.f,0.f,0.f,0.f,0.f,0.f,0.f};
        #pragma unroll
        for (int kc = 0; kc < 4; ++kc)
            acc0 = __builtin_amdgcn_wmma_f32_16x16x32_bf16(
                       false, a[kc].bf, false, b0[kc].bf, (short)0, acc0,
                       false, false);
        #pragma unroll
        for (int kc = 0; kc < 4; ++kc)
            acc1 = __builtin_amdgcn_wmma_f32_16x16x32_bf16(
                       false, a[kc].bf, false, b1[kc].bf, (short)0, acc1,
                       false, false);

        #pragma unroll
        for (int r8 = 0; r8 < 8; ++r8) {
            size_t rbase = (size_t)(m0 + r8 + half * 8) * 1024;
            y[rbase + ncol0] = acc0[r8] + bias0;
            y[rbase + ncol1] = acc1[r8] + bias1;
        }
    }
}

// ---------------------------------------------------------------------------
// LayerNorm over D=1024, in place on y (= d_out). One block per row.
// ---------------------------------------------------------------------------
__global__ void __launch_bounds__(256) ln_kernel(
    float* __restrict__ y, const float* __restrict__ g,
    const float* __restrict__ bvec)
{
    const int tid = threadIdx.x;
    const size_t row = blockIdx.x;
    float* rp = y + row * 1024;
    float4 v = reinterpret_cast<const float4*>(rp)[tid];
    float s = v.x + v.y + v.z + v.w;
    float q = v.x*v.x + v.y*v.y + v.z*v.z + v.w*v.w;
    #pragma unroll
    for (int off = 16; off >= 1; off >>= 1) {
        s += __shfl_down(s, off, 32);
        q += __shfl_down(q, off, 32);
    }
    __shared__ float rs[8], rq[8];
    __shared__ float mu_s, inv_s;
    if ((tid & 31) == 0) { rs[tid >> 5] = s; rq[tid >> 5] = q; }
    __syncthreads();
    if (tid == 0) {
        float S = 0.f, Q = 0.f;
        #pragma unroll
        for (int i = 0; i < 8; ++i) { S += rs[i]; Q += rq[i]; }
        float mu  = S * (1.0f / 1024.0f);
        float var = Q * (1.0f / 1024.0f) - mu * mu;
        mu_s  = mu;
        inv_s = rsqrtf(var + 1e-5f);
    }
    __syncthreads();
    float mu = mu_s, inv = inv_s;
    float4 gg = reinterpret_cast<const float4*>(g)[tid];
    float4 bb = reinterpret_cast<const float4*>(bvec)[tid];
    float4 o;
    o.x = (v.x - mu) * inv * gg.x + bb.x;
    o.y = (v.y - mu) * inv * gg.y + bb.y;
    o.z = (v.z - mu) * inv * gg.z + bb.z;
    o.w = (v.w - mu) * inv * gg.w + bb.w;
    reinterpret_cast<float4*>(rp)[tid] = o;
}

// ---------------------------------------------------------------------------
extern "C" void kernel_launch(void* const* d_in, const int* in_sizes, int n_in,
                              void* d_out, int out_size, void* d_ws, size_t ws_size,
                              hipStream_t stream)
{
    (void)in_sizes; (void)n_in; (void)out_size; (void)ws_size;

    const float* x     = (const float*)d_in[0];
    const float* alpha = (const float*)d_in[1];
    const float* omega = (const float*)d_in[2];
    const float* W_in  = (const float*)d_in[3];
    const float* R     = (const float*)d_in[4];
    const float* W_out = (const float*)d_in[5];
    const float* b_out = (const float*)d_in[6];
    const float* ln_g  = (const float*)d_in[7];
    const float* ln_b  = (const float*)d_in[8];

    float* out = (float*)d_out;                       // (B,S,D)
    float* srf = out + (size_t)8 * 4096 * 1024;       // (B,K)
    float* sif = srf + 8 * 128;                       // (B,K)

    // workspace layout
    char* ws = (char*)d_ws;
    float*          beta  = (float*)ws;                                    // 16 MB
    float*          srseq = (float*)(ws + ((size_t)16 << 20));             // 16 MB
    unsigned short* Bp1   = (unsigned short*)(ws + ((size_t)32 << 20));    // 256 KB
    unsigned short* Bp2   = (unsigned short*)(ws + ((size_t)32 << 20) + ((size_t)256 << 10));

    // pack weights into WMMA B-operand bf16 layout
    pack_b_kernel<<<512, 256, 0, stream>>>(W_in,  Bp1, 128, 1024);   // 32 ksteps
    pack_b_kernel<<<512, 256, 0, stream>>>(W_out, Bp2, 1024, 128);   // 4 ksteps

    // beta = x @ W_in^T
    gemm_beta_kernel<<<256, 256, 0, stream>>>(x, (const uint4*)Bp1, beta);

    // recurrent scan (fp32), also emits sr_f / si_f
    scan_kernel<<<8, 512, 0, stream>>>(beta, alpha, omega, R, srseq, srf, sif);

    // y = srseq @ W_out^T + b_out   (into d_out)
    gemm_out_kernel<<<256, 256, 0, stream>>>(srseq, (const uint4*)Bp2, b_out, out);

    // LayerNorm in place on d_out
    ln_kernel<<<32768, 256, 0, stream>>>(out, ln_g, ln_b);
}
